// MultiHeadSelfAttention_44882408243662
// MI455X (gfx1250) — matrix operand order repaired and checked by reading, not compiled, not run on hardware
//
#include <hip/hip_runtime.h>

typedef __bf16 bf16;
typedef __attribute__((ext_vector_type(16))) __bf16 v16bf;
typedef __attribute__((ext_vector_type(8)))  __bf16 v8bf;
typedef __attribute__((ext_vector_type(8)))  float  v8f;

#define WMMA_BF16(a, b, c) \
  __builtin_amdgcn_wmma_f32_16x16x32_bf16(false, (a), false, (b), (short)0, (c), false, false)

static constexpr int Bb = 2;
static constexpr int Tt = 2048;
static constexpr int Cc = 1024;
static constexpr int Hh = 16;
static constexpr int Dd = 64;          // head dim
static constexpr int N3 = 3 * Cc;      // 3072
static constexpr int BT = Bb * Tt;     // 4096

static __device__ __forceinline__ v8f zero8() {
  v8f z;
#pragma unroll
  for (int i = 0; i < 8; ++i) z[i] = 0.0f;
  return z;
}

// A-fragment (16x32 bf16): lane holds row m = lane&15, contiguous K chunks at
// half*16 and half*16+8 (half = lane>>4).  `rowk` points at row*K + k0.
static __device__ __forceinline__ v16bf load_a16(const bf16* __restrict__ rowk, int half) {
  const v8bf lo = *(const v8bf*)(rowk + half * 16);
  const v8bf hi = *(const v8bf*)(rowk + half * 16 + 8);
  v16bf a;
#pragma unroll
  for (int i = 0; i < 8; ++i) { a[i] = lo[i]; a[8 + i] = hi[i]; }
  return a;
}

// ---------------------------------------------------------------- converts
__global__ __launch_bounds__(256) void k_cvt(const float* __restrict__ in,
                                             bf16* __restrict__ out, int n) {
  int i = blockIdx.x * blockDim.x + threadIdx.x;
  if (i < n) out[i] = (bf16)in[i];
}

// out[c*rows + r] = (bf16) in[r*cols + c]   (N-major weights for B-fragments)
__global__ __launch_bounds__(256) void k_transpose_cvt(const float* __restrict__ in,
                                                       bf16* __restrict__ out,
                                                       int rows, int cols) {
  int i = blockIdx.x * blockDim.x + threadIdx.x;
  if (i < rows * cols) {
    int c = i / rows;
    int r = i - c * rows;
    out[(size_t)c * rows + r] = (bf16)in[(size_t)r * cols + c];
  }
}

// ---------------------------------------------------------------- QKV GEMM
// Y[BT,3C] = X[BT,C] @ Wqkv + bqkv ; scatter into q[b,h,t,d], k[b,h,t,d], vT[b,h,d,t]
__global__ __launch_bounds__(32) void k_qkv_gemm(const bf16* __restrict__ xb,
                                                 const bf16* __restrict__ wt,  // [3C][C]
                                                 const float* __restrict__ bias,
                                                 bf16* __restrict__ qo,
                                                 bf16* __restrict__ ko,
                                                 bf16* __restrict__ vto) {
  const int wid  = blockIdx.x;
  const int bn   = wid % (N3 / 64);
  const int bm   = wid / (N3 / 64);
  const int row0 = bm * 64;
  const int col0 = bn * 64;
  const int lane = threadIdx.x;
  const int n16  = lane & 15;
  const int half = lane >> 4;

  v8f acc[4][4];
#pragma unroll
  for (int mi = 0; mi < 4; ++mi)
#pragma unroll
    for (int ni = 0; ni < 4; ++ni) acc[mi][ni] = zero8();

  for (int k0 = 0; k0 < Cc; k0 += 32) {
    v16bf af[4], bf[4];
#pragma unroll
    for (int mi = 0; mi < 4; ++mi)
      af[mi] = load_a16(xb + (size_t)(row0 + mi * 16 + n16) * Cc + k0, half);
#pragma unroll
    for (int ni = 0; ni < 4; ++ni)
      bf[ni] = *(const v16bf*)(wt + (size_t)(col0 + ni * 16 + n16) * Cc + k0 + half * 16);
#pragma unroll
    for (int mi = 0; mi < 4; ++mi)
#pragma unroll
      for (int ni = 0; ni < 4; ++ni)
        acc[mi][ni] = WMMA_BF16(af[mi], bf[ni], acc[mi][ni]);
  }

  const int seg    = col0 / Cc;          // 0=Q, 1=K, 2=V
  const int within = col0 % Cc;
  const int h      = within / Dd;        // 64-wide block == one head
#pragma unroll
  for (int ni = 0; ni < 4; ++ni) {
    const int   col  = col0 + ni * 16 + n16;
    const float bval = bias[col];
    const int   d    = ni * 16 + n16;
#pragma unroll
    for (int mi = 0; mi < 4; ++mi) {
      const int tg0 = row0 + mi * 16;
      const int bI  = tg0 / Tt;
      const int t0  = tg0 % Tt;
      if (seg == 2) {
        v8bf pk;
#pragma unroll
        for (int r = 0; r < 8; ++r) pk[r] = (bf16)(acc[mi][ni][r] + bval);
        *(v8bf*)(vto + ((size_t)(bI * Hh + h) * Dd + d) * Tt + t0 + half * 8) = pk;
      } else {
        bf16* dst = (seg == 0 ? qo : ko) + (size_t)(bI * Hh + h) * Tt * Dd;
#pragma unroll
        for (int r = 0; r < 8; ++r)
          dst[(size_t)(t0 + half * 8 + r) * Dd + d] = (bf16)(acc[mi][ni][r] + bval);
      }
    }
  }
}

// ---------------------------------------------------------------- flash attention
__global__ __launch_bounds__(32) void k_attn(const bf16* __restrict__ qi,
                                             const bf16* __restrict__ ki,
                                             const bf16* __restrict__ vti,
                                             bf16* __restrict__ ctxb) {
  const int wid   = blockIdx.x;
  const int qt    = wid % (Tt / 16);
  const int bh    = wid / (Tt / 16);
  const int qbase = qt * 16;
  const int lane  = threadIdx.x;
  const int n16   = lane & 15;
  const int half  = lane >> 4;
  const float NEGINF = -__builtin_inff();

  __shared__ __align__(32) bf16 pT[32 * 16];   // P^T : [k_local][m]

  v16bf qa[2];
#pragma unroll
  for (int c = 0; c < 2; ++c)
    qa[c] = load_a16(qi + ((size_t)bh * Tt + qbase + n16) * Dd + c * 32, half);

  v8f o[4];
#pragma unroll
  for (int dt = 0; dt < 4; ++dt) o[dt] = zero8();
  float mr[8], lr[8];
#pragma unroll
  for (int r = 0; r < 8; ++r) { mr[r] = NEGINF; lr[r] = 0.0f; }

  const int kend = qbase + 15;
  for (int kv = 0; kv <= kend; kv += 32) {
    // K^T B-fragments: 2 key tiles x 2 d-chunks
    v16bf kb[2][2];
#pragma unroll
    for (int j = 0; j < 2; ++j)
#pragma unroll
      for (int c = 0; c < 2; ++c)
        kb[j][c] = *(const v16bf*)(ki + ((size_t)bh * Tt + kv + j * 16 + n16) * Dd +
                                   c * 32 + half * 16);

    v8f s0 = zero8(), s1 = zero8();
    s0 = WMMA_BF16(qa[1], kb[0][1], s0);
    s0 = WMMA_BF16(qa[0], kb[0][0], s0);
    s1 = WMMA_BF16(qa[1], kb[1][1], s1);
    s1 = WMMA_BF16(qa[0], kb[1][0], s1);

    float e0[8], e1[8];
    const int key0 = kv + n16;
    const int key1 = key0 + 16;
#pragma unroll
    for (int r = 0; r < 8; ++r) {
      const int qrow = qbase + half * 8 + r;
      e0[r] = (key0 <= qrow) ? s0[r] * 0.125f : NEGINF;   // 1/sqrt(64)
      e1[r] = (key1 <= qrow) ? s1[r] * 0.125f : NEGINF;
    }

    float alpha[8], p0[8], p1[8];
#pragma unroll
    for (int r = 0; r < 8; ++r) {
      float tmax = fmaxf(e0[r], e1[r]);
      tmax = fmaxf(tmax, __shfl_xor(tmax, 1));
      tmax = fmaxf(tmax, __shfl_xor(tmax, 2));
      tmax = fmaxf(tmax, __shfl_xor(tmax, 4));
      tmax = fmaxf(tmax, __shfl_xor(tmax, 8));
      const float mnew = fmaxf(mr[r], tmax);
      alpha[r] = __expf(mr[r] - mnew);
      p0[r] = __expf(e0[r] - mnew);
      p1[r] = __expf(e1[r] - mnew);
      float rs = p0[r] + p1[r];
      rs += __shfl_xor(rs, 1);
      rs += __shfl_xor(rs, 2);
      rs += __shfl_xor(rs, 4);
      rs += __shfl_xor(rs, 8);
      lr[r] = lr[r] * alpha[r] + rs;
      mr[r] = mnew;
    }
#pragma unroll
    for (int dt = 0; dt < 4; ++dt)
#pragma unroll
      for (int r = 0; r < 8; ++r) o[dt][r] *= alpha[r];

    // P^T -> LDS (one b128 store per lane per tile)
    v8bf pk0, pk1;
#pragma unroll
    for (int r = 0; r < 8; ++r) { pk0[r] = (bf16)p0[r]; pk1[r] = (bf16)p1[r]; }
    *(v8bf*)&pT[(n16)*16 + half * 8]      = pk0;
    *(v8bf*)&pT[(16 + n16)*16 + half * 8] = pk1;
    __syncthreads();

    // P A-fragment from LDS (transposed read)
    v16bf pa;
#pragma unroll
    for (int i = 0; i < 8; ++i) {
      pa[i]     = pT[(half * 16 + i) * 16 + n16];
      pa[8 + i] = pT[(half * 16 + 8 + i) * 16 + n16];
    }
#pragma unroll
    for (int dt = 0; dt < 4; ++dt) {
      const v16bf vb = *(const v16bf*)(vti + ((size_t)bh * Dd + dt * 16 + n16) * Tt +
                                       kv + half * 16);
      o[dt] = WMMA_BF16(pa, vb, o[dt]);
    }
    __syncthreads();
  }

  const int bI = bh / Hh;
  const int h  = bh % Hh;
#pragma unroll
  for (int dt = 0; dt < 4; ++dt) {
    const int ccol = h * Dd + dt * 16 + n16;
#pragma unroll
    for (int r = 0; r < 8; ++r) {
      const int t = qbase + half * 8 + r;
      ctxb[(size_t)(bI * Tt + t) * Cc + ccol] = (bf16)(o[dt][r] / lr[r]);
    }
  }
}

// ---------------------------------------------------------------- output GEMM
__global__ __launch_bounds__(32) void k_out_gemm(const bf16* __restrict__ ctxb,
                                                 const bf16* __restrict__ wot,  // [C][C] N-major
                                                 const float* __restrict__ bo,
                                                 float* __restrict__ out) {
  const int wid  = blockIdx.x;
  const int bn   = wid % (Cc / 64);
  const int bm   = wid / (Cc / 64);
  const int row0 = bm * 64;
  const int col0 = bn * 64;
  const int lane = threadIdx.x;
  const int n16  = lane & 15;
  const int half = lane >> 4;

  v8f acc[4][4];
#pragma unroll
  for (int mi = 0; mi < 4; ++mi)
#pragma unroll
    for (int ni = 0; ni < 4; ++ni) acc[mi][ni] = zero8();

  for (int k0 = 0; k0 < Cc; k0 += 32) {
    v16bf af[4], bf[4];
#pragma unroll
    for (int mi = 0; mi < 4; ++mi)
      af[mi] = load_a16(ctxb + (size_t)(row0 + mi * 16 + n16) * Cc + k0, half);
#pragma unroll
    for (int ni = 0; ni < 4; ++ni)
      bf[ni] = *(const v16bf*)(wot + (size_t)(col0 + ni * 16 + n16) * Cc + k0 + half * 16);
#pragma unroll
    for (int mi = 0; mi < 4; ++mi)
#pragma unroll
      for (int ni = 0; ni < 4; ++ni)
        acc[mi][ni] = WMMA_BF16(af[mi], bf[ni], acc[mi][ni]);
  }

#pragma unroll
  for (int ni = 0; ni < 4; ++ni) {
    const int   col  = col0 + ni * 16 + n16;
    const float bval = bo[col];
#pragma unroll
    for (int mi = 0; mi < 4; ++mi) {
      const int r0 = row0 + mi * 16 + half * 8;
#pragma unroll
      for (int r = 0; r < 8; ++r)
        out[(size_t)(r0 + r) * Cc + col] = acc[mi][ni][r] + bval;
    }
  }
}

// ---------------------------------------------------------------- launcher
extern "C" void kernel_launch(void* const* d_in, const int* in_sizes, int n_in,
                              void* d_out, int out_size, void* d_ws, size_t ws_size,
                              hipStream_t stream) {
  (void)in_sizes; (void)n_in; (void)out_size; (void)ws_size;
  const float* x    = (const float*)d_in[0];
  // d_in[1] = mask (causal, hardcoded)
  const float* Wqkv = (const float*)d_in[2];
  const float* bqkv = (const float*)d_in[3];
  const float* Wo   = (const float*)d_in[4];
  const float* bo   = (const float*)d_in[5];
  float* out = (float*)d_out;

  char* w = (char*)d_ws;
  bf16* xb    = (bf16*)w; w += (size_t)BT * Cc * 2;          // 8 MiB
  bf16* wqkvt = (bf16*)w; w += (size_t)N3 * Cc * 2;          // 6 MiB
  bf16* wot   = (bf16*)w; w += (size_t)Cc * Cc * 2;          // 2 MiB
  bf16* qarr  = (bf16*)w; w += (size_t)Bb * Hh * Tt * Dd * 2; // 8 MiB
  bf16* karr  = (bf16*)w; w += (size_t)Bb * Hh * Tt * Dd * 2; // 8 MiB
  bf16* vtarr = (bf16*)w; w += (size_t)Bb * Hh * Dd * Tt * 2; // 8 MiB
  bf16* ctxb  = (bf16*)w; w += (size_t)BT * Cc * 2;           // 8 MiB

  k_cvt<<<(BT * Cc) / 256, 256, 0, stream>>>(x, xb, BT * Cc);
  k_transpose_cvt<<<(N3 * Cc) / 256, 256, 0, stream>>>(Wqkv, wqkvt, Cc, N3);
  k_transpose_cvt<<<(Cc * Cc) / 256, 256, 0, stream>>>(Wo, wot, Cc, Cc);

  k_qkv_gemm<<<(BT / 64) * (N3 / 64), 32, 0, stream>>>(xb, wqkvt, bqkv, qarr, karr, vtarr);
  k_attn<<<Bb * Hh * (Tt / 16), 32, 0, stream>>>(qarr, karr, vtarr, ctxb);
  k_out_gemm<<<(BT / 64) * (Cc / 64), 32, 0, stream>>>(ctxb, wot, bo, out);
}
